// StockLevelFactorLearning_80762565034572
// MI455X (gfx1250) — matrix operand order repaired and checked by reading, not compiled, not run on hardware
//
#include <hip/hip_runtime.h>
#include <hip/hip_bf16.h>
#include <math.h>

// ---------------------------------------------------------------------------
// CDNA5 (gfx1250) wave32 WMMA types
// ---------------------------------------------------------------------------
typedef __attribute__((ext_vector_type(16))) __bf16 v16bf;
typedef __attribute__((ext_vector_type(8)))  float  v8f;

union FragB16 { v16bf v; unsigned short us[16]; uint4 q[2]; };
union FragAcc { v8f v; float f[8]; };

__device__ __forceinline__ unsigned short f2bf(float f) {
  unsigned int u = __float_as_uint(f);
  u += 0x7FFFu + ((u >> 16) & 1u);   // round-to-nearest-even
  return (unsigned short)(u >> 16);
}
__device__ __forceinline__ float bf2f(unsigned short h) {
  return __uint_as_float(((unsigned int)h) << 16);
}

// ---------------------------------------------------------------------------
// Kernel 0: split prices (f32) into planar bf16 hi/lo arrays, one pass.
// ---------------------------------------------------------------------------
__global__ __launch_bounds__(256) void split_prices_kernel(
    const float* __restrict__ P,
    unsigned short* __restrict__ Phi,
    unsigned short* __restrict__ Plo,
    size_t n)
{
  const size_t base = ((size_t)blockIdx.x * 256 + threadIdx.x) * 8;
  if (base >= n) return;
  float4 x0 = *(const float4*)(P + base);
  float4 x1 = *(const float4*)(P + base + 4);
  float xs[8] = {x0.x, x0.y, x0.z, x0.w, x1.x, x1.y, x1.z, x1.w};
  unsigned int h[8], l[8];
  #pragma unroll
  for (int e = 0; e < 8; ++e) {
    h[e] = f2bf(xs[e]);
    l[e] = f2bf(xs[e] - bf2f((unsigned short)h[e]));
  }
  uint4 hq, lq;
  hq.x = h[0] | (h[1] << 16); hq.y = h[2] | (h[3] << 16);
  hq.z = h[4] | (h[5] << 16); hq.w = h[6] | (h[7] << 16);
  lq.x = l[0] | (l[1] << 16); lq.y = l[2] | (l[3] << 16);
  lq.z = l[4] | (l[5] << 16); lq.w = l[6] | (l[7] << 16);
  *(uint4*)(Phi + base) = hq;
  *(uint4*)(Plo + base) = lq;
}

// ---------------------------------------------------------------------------
// Kernel 1: M[i,j] = softmax_j( E_i . E_j, diag=-inf ) * beta[i,j] (diag=0)
// stored as split bf16 (hi, lo).  One workgroup (256 threads) per row i.
// ---------------------------------------------------------------------------
__global__ __launch_bounds__(256) void build_M_kernel(
    const float* __restrict__ E, const float* __restrict__ beta,
    unsigned short* __restrict__ Mhi, unsigned short* __restrict__ Mlo,
    int I, int D)
{
  extern __shared__ float smem[];
  float* ei  = smem;           // D floats
  float* w   = smem + D;       // I floats
  float* red = smem + D + I;   // 256 floats

  const int i   = blockIdx.x;
  const int tid = threadIdx.x;

  for (int d = tid; d < D; d += 256) ei[d] = E[(size_t)i * D + d];
  __syncthreads();

  for (int j = tid; j < I; j += 256) {
    const float* Ej = E + (size_t)j * D;
    float s = 0.f;
    #pragma unroll 4
    for (int d = 0; d < D; d += 4) {
      float4 a = *(const float4*)(ei + d);
      float4 b = *(const float4*)(Ej + d);
      s = fmaf(a.x, b.x, s); s = fmaf(a.y, b.y, s);
      s = fmaf(a.z, b.z, s); s = fmaf(a.w, b.w, s);
    }
    w[j] = (j == i) ? -INFINITY : s;
  }
  __syncthreads();

  float m = -INFINITY;
  for (int j = tid; j < I; j += 256) m = fmaxf(m, w[j]);
  red[tid] = m; __syncthreads();
  for (int s = 128; s > 0; s >>= 1) {
    if (tid < s) red[tid] = fmaxf(red[tid], red[tid + s]);
    __syncthreads();
  }
  const float rowmax = red[0];
  __syncthreads();

  float psum = 0.f;
  for (int j = tid; j < I; j += 256) {
    float ev = (j == i) ? 0.f : __expf(w[j] - rowmax);
    w[j] = ev;
    psum += ev;
  }
  red[tid] = psum; __syncthreads();
  for (int s = 128; s > 0; s >>= 1) {
    if (tid < s) red[tid] += red[tid + s];
    __syncthreads();
  }
  const float inv = 1.f / red[0];

  for (int j = tid; j < I; j += 256) {
    float mv = (j == i) ? 0.f : w[j] * inv * beta[(size_t)i * I + j];
    unsigned short h = f2bf(mv);
    unsigned short l = f2bf(mv - bf2f(h));
    Mhi[(size_t)i * I + j] = h;
    Mlo[(size_t)i * I + j] = l;
  }
}

// ---------------------------------------------------------------------------
// Kernel 2: V = P @ M^T using split-bf16 fp32 emulation (hi*hi + hi*lo + lo*hi,
// f32 accumulators).  All operands are pre-split packed bf16, so the inner
// loop is pure b128 loads + WMMA.  128x128 C tile per 256-thread block;
// 8 wave32s in a 2x4 grid; each wave owns 64x32 = 4x2 tiles of 16x16.
// K-loop indexed off loop-invariant base pointers and unrolled 4x so the
// k-advance folds into the load immediate offsets.
// ---------------------------------------------------------------------------
__global__ __launch_bounds__(256) void vrp_gemm_kernel(
    const unsigned short* __restrict__ Phi,
    const unsigned short* __restrict__ Plo,
    const unsigned short* __restrict__ Mhi,
    const unsigned short* __restrict__ Mlo,
    float* __restrict__ V,
    int T, int I)
{
  const int lane = threadIdx.x & 31;
  const int wave = threadIdx.x >> 5;
  const int wm = wave >> 2;             // 0..1 (rows of 64)
  const int wn = wave & 3;              // 0..3 (cols of 32)
  const int t0 = blockIdx.y * 128 + wm * 64;
  const int i0 = blockIdx.x * 128 + wn * 32;

  const int nlane   = lane & 15;        // A: M index, B/C: N index
  const int halfsel = lane >> 4;        // lane half
  const int abase   = halfsel * 16;     // A K base: 0-15 or 16-31
  const int bkbase  = halfsel * 16;     // B K base: 0-15 or 16-31

  FragAcc acc[4][2];
  #pragma unroll
  for (int mt = 0; mt < 4; ++mt)
    #pragma unroll
    for (int nt = 0; nt < 2; ++nt)
      #pragma unroll
      for (int v = 0; v < 8; ++v) acc[mt][nt].f[v] = 0.f;

  // Loop-invariant per-lane base pointers.
  const unsigned short* pAhi[4];
  const unsigned short* pAlo[4];
  const unsigned short* pBhi[2];
  const unsigned short* pBlo[2];
  #pragma unroll
  for (int mt = 0; mt < 4; ++mt) {
    const size_t off = (size_t)(t0 + mt * 16 + nlane) * I + abase;
    pAhi[mt] = Phi + off;
    pAlo[mt] = Plo + off;
  }
  #pragma unroll
  for (int nt = 0; nt < 2; ++nt) {
    const size_t off = (size_t)(i0 + nt * 16 + nlane) * I + bkbase;
    pBhi[nt] = Mhi + off;
    pBlo[nt] = Mlo + off;
  }

  const int nsteps = I >> 5;            // K = I in chunks of 32
  #pragma unroll 4
  for (int kk = 0; kk < nsteps; ++kk) {
    const int koff = kk << 5;           // element offset along K

    // ---- B fragments (M matrix, already bf16): 2 x b128 each (contiguous)
    FragB16 bhi[2], blo[2];
    #pragma unroll
    for (int nt = 0; nt < 2; ++nt) {
      const uint4* ph = (const uint4*)(pBhi[nt] + koff);
      const uint4* pl = (const uint4*)(pBlo[nt] + koff);
      bhi[nt].q[0] = ph[0]; bhi[nt].q[1] = ph[1];
      blo[nt].q[0] = pl[0]; blo[nt].q[1] = pl[1];
    }

    // ---- A fragments (prices, pre-split bf16): 2 x b128 each (contiguous)
    FragB16 ahi[4], alo[4];
    #pragma unroll
    for (int mt = 0; mt < 4; ++mt) {
      const uint4* ph = (const uint4*)(pAhi[mt] + koff);
      const uint4* pl = (const uint4*)(pAlo[mt] + koff);
      ahi[mt].q[0] = ph[0]; ahi[mt].q[1] = ph[1];
      alo[mt].q[0] = pl[0]; alo[mt].q[1] = pl[1];
    }

    // ---- 24 WMMAs: hi*hi + hi*lo + lo*hi per 16x16 tile
    #pragma unroll
    for (int mt = 0; mt < 4; ++mt)
      #pragma unroll
      for (int nt = 0; nt < 2; ++nt) {
        acc[mt][nt].v = __builtin_amdgcn_wmma_f32_16x16x32_bf16(
            false, ahi[mt].v, false, bhi[nt].v, (short)0, acc[mt][nt].v, false, false);
        acc[mt][nt].v = __builtin_amdgcn_wmma_f32_16x16x32_bf16(
            false, ahi[mt].v, false, blo[nt].v, (short)0, acc[mt][nt].v, false, false);
        acc[mt][nt].v = __builtin_amdgcn_wmma_f32_16x16x32_bf16(
            false, alo[mt].v, false, bhi[nt].v, (short)0, acc[mt][nt].v, false, false);
      }
  }

  // ---- epilogue: write V only (C/D layout: VGPR v -> row v or v+8)
  #pragma unroll
  for (int mt = 0; mt < 4; ++mt)
    #pragma unroll
    for (int nt = 0; nt < 2; ++nt) {
      #pragma unroll
      for (int v = 0; v < 8; ++v) {
        const int m = v + halfsel * 8;
        const int t = t0 + mt * 16 + m;
        const int i = i0 + nt * 16 + nlane;
        V[(size_t)t * I + i] = acc[mt][nt].f[v];
      }
    }
}

// ---------------------------------------------------------------------------
// Kernel 3: U = V - P (overwrites the bf16 scratch that lived in the U half)
// ---------------------------------------------------------------------------
__global__ __launch_bounds__(256) void u_kernel(
    const float* __restrict__ V, const float* __restrict__ P,
    float* __restrict__ U, size_t n)
{
  const size_t base = ((size_t)blockIdx.x * 256 + threadIdx.x) * 4;
  if (base >= n) return;
  float4 v = *(const float4*)(V + base);
  float4 p = *(const float4*)(P + base);
  float4 u;
  u.x = v.x - p.x; u.y = v.y - p.y; u.z = v.z - p.z; u.w = v.w - p.w;
  *(float4*)(U + base) = u;
}

// ---------------------------------------------------------------------------
// Host launch
// ---------------------------------------------------------------------------
extern "C" void kernel_launch(void* const* d_in, const int* in_sizes, int n_in,
                              void* d_out, int out_size, void* d_ws, size_t ws_size,
                              hipStream_t stream) {
  const float* prices = (const float*)d_in[0];   // [T, I]
  const float* E      = (const float*)d_in[1];   // [I, D]
  const float* beta   = (const float*)d_in[2];   // [I, I]

  const int I = (int)(sqrt((double)in_sizes[2]) + 0.5);
  const int D = in_sizes[1] / I;
  const int T = in_sizes[0] / I;
  const size_t TI = (size_t)T * I;

  // Workspace: split M (bf16 hi/lo), 4 bytes per element of beta.
  unsigned short* Mhi = (unsigned short*)d_ws;
  unsigned short* Mlo = Mhi + (size_t)I * I;

  // d_out: [ V (T*I f32) | U (T*I f32) ].  The U half doubles as scratch for
  // the pre-split prices (Phi: T*I bf16, Plo: T*I bf16 — exactly T*I*4 bytes)
  // until the final u_kernel overwrites it with U = V - P.
  float* V = (float*)d_out;
  unsigned short* Phi = (unsigned short*)(V + TI);
  unsigned short* Plo = Phi + TI;
  float* U = V + TI;

  // 0) split prices to planar bf16 hi/lo (scratch in U half)
  split_prices_kernel<<<(unsigned)((TI / 8 + 255) / 256), 256, 0, stream>>>(
      prices, Phi, Plo, TI);

  // 1) attention-weighted beta matrix, split bf16
  const size_t smem = (size_t)(D + I + 256) * sizeof(float);
  build_M_kernel<<<I, 256, smem, stream>>>(E, beta, Mhi, Mlo, I, D);

  // 2) V = P @ M^T  (WMMA, split-bf16 fp32 emulation)
  dim3 grid(I / 128, T / 128);
  vrp_gemm_kernel<<<grid, 256, 0, stream>>>(Phi, Plo, Mhi, Mlo, V, T, I);

  // 3) U = V - P  (overwrites scratch)
  u_kernel<<<(unsigned)((TI / 4 + 255) / 256), 256, 0, stream>>>(
      V, prices, U, TI);
}
